// NMF_42992622633213
// MI455X (gfx1250) — matrix operand order repaired and checked by reading, not compiled, not run on hardware
//
#include <hip/hip_runtime.h>
#include <hip/hip_bf16.h>

typedef __attribute__((ext_vector_type(16))) _Float16 v16h;
typedef __attribute__((ext_vector_type(8)))  float    v8f;

#define NUSER 512
#define NITEM 1024

// ---------------------------------------------------------------------------
// Kernel 1: O(U+I) precompute of the bilinear decomposition.
//   Au[u][j] = user_emb_mlp[uid]  . W1[j, 0:64]
//   Bi[i][j] = item_emb_mlp[iid]  . W1[j, 64:128]
//   Su[u]    = user_emb_mf[uid]   . Wo[0:64]
//   Si[i]    = item_emb_mf[iid]   . Wo[64:128]
// ---------------------------------------------------------------------------
__global__ void __launch_bounds__(64)
nmf_precompute(const int* __restrict__ user_id,
               const int* __restrict__ item_id,
               const float* __restrict__ user_emb_mf,
               const float* __restrict__ item_emb_mf,
               const float* __restrict__ user_emb_mlp,
               const float* __restrict__ item_emb_mlp,
               const float* __restrict__ W1,
               const float* __restrict__ Wo,
               float* __restrict__ Au, float* __restrict__ Bi,
               float* __restrict__ Su, float* __restrict__ Si)
{
    __shared__ float red[64];
    const int b = blockIdx.x;
    const int j = threadIdx.x;
    if (b < NUSER) {
        const int uid = user_id[b];
        const float* e = user_emb_mlp + uid * 64;
        const float* w = W1 + j * 128;            // left half of W1 row j
        float a = 0.f;
        #pragma unroll
        for (int k = 0; k < 64; k++) a += e[k] * w[k];
        Au[b * 64 + j] = a;
        red[j] = user_emb_mf[uid * 64 + j] * Wo[j];
        __syncthreads();
        if (j == 0) { float s = 0.f; for (int k = 0; k < 64; k++) s += red[k]; Su[b] = s; }
    } else {
        const int n = b - NUSER;
        const int iid = item_id[n];
        const float* e = item_emb_mlp + iid * 64;
        const float* w = W1 + j * 128 + 64;       // right half of W1 row j
        float a = 0.f;
        #pragma unroll
        for (int k = 0; k < 64; k++) a += e[k] * w[k];
        Bi[n * 64 + j] = a;
        red[j] = item_emb_mf[iid * 64 + j] * Wo[64 + j];
        __syncthreads();
        if (j == 0) { float s = 0.f; for (int k = 0; k < 64; k++) s += red[k]; Si[n] = s; }
    }
}

// ---------------------------------------------------------------------------
// Kernel 2: per-pair MLP via WMMA. 1 block = 1 user, 4 waves; each wave owns
// 256 items as 16 tiles of 16 pairs. Bi tiles are streamed into wave-private
// LDS ping-pong buffers with global_load_async_to_lds_b128 (ASYNCcnt
// pipelining, in-order completion => s_wait_asynccnt 8 retires the older
// tile while the newer one is in flight). No cross-wave barriers needed.
// ---------------------------------------------------------------------------
__global__ void __launch_bounds__(128)
nmf_main(const float* __restrict__ Au, const float* __restrict__ Bi,
         const float* __restrict__ Su, const float* __restrict__ Si,
         const float* __restrict__ b1, const float* __restrict__ W2,
         const float* __restrict__ b2, const float* __restrict__ W4,
         const float* __restrict__ b4, const float* __restrict__ Wo,
         const float* __restrict__ bo, float* __restrict__ out)
{
    __shared__ __align__(16) float ldsBi[4][2][16 * 64]; // per-wave ping-pong (2 x 4 KB)
    __shared__ __align__(16) float ldsH2[4][16 * 32];    // per-wave H2 tile (2 KB)

    const int u    = blockIdx.x;
    const int wave = threadIdx.x >> 5;
    const int lane = threadIdx.x & 31;
    const int hi   = lane >> 4;   // half-wave id
    const int ln   = lane & 15;

    // ---- loop-invariant B fragments (16-bit B layout: N = ln, K = (t/8)*16 + hi*8 + t%8) ----
    v16h bf[4]; // [kc*2+nc]: W2^T chunk (K = kc*32.., N = nc*16..)
    #pragma unroll
    for (int kc = 0; kc < 2; kc++) {
        #pragma unroll
        for (int nc = 0; nc < 2; nc++) {
            const float* src = W2 + (nc * 16 + ln) * 64 + kc * 32 + hi * 8;
            v16h tmp;
            #pragma unroll
            for (int t = 0; t < 16; t++) tmp[t] = (_Float16)src[((t >> 3) << 4) + (t & 7)];
            bf[kc * 2 + nc] = tmp;
        }
    }
    v16h b4f; // W4^T (32x16)
    {
        const float* src = W4 + ln * 32 + hi * 8;
        #pragma unroll
        for (int t = 0; t < 16; t++) b4f[t] = (_Float16)src[((t >> 3) << 4) + (t & 7)];
    }

    // ---- loop-invariant A-side partial: Au[u][k] + b1[k] for this lane's K set ----
    // A layout (16-bit, 16x32): lane holds row M = ln; half t -> K = c*32 + (t/8)*16 + hi*8 + t%8
    const float* AuRow = Au + u * 64;
    float av[32];
    #pragma unroll
    for (int c = 0; c < 2; c++)
        #pragma unroll
        for (int t = 0; t < 16; t++) {
            const int k = c * 32 + ((t >> 3) << 4) + (hi << 3) + (t & 7);
            av[c * 16 + t] = AuRow[k] + b1[k];
        }

    const float su     = Su[u];
    const float bias2a = b2[ln];
    const float bias2b = b2[16 + ln];
    const float bias4  = b4[ln];
    const float woh    = Wo[128 + ln];
    const float bov    = bo[0];

    const int itemBase0 = wave * 256;

    // LDS byte offsets (generic-pointer low 32 bits == wave-relative LDS addr)
    const unsigned biBase0 = (unsigned)(size_t)(const void*)&ldsBi[wave][0][0];
    const unsigned biBase1 = (unsigned)(size_t)(const void*)&ldsBi[wave][1][0];

    // async-stage one 16x64 f32 tile (32 lanes x 8 x b128 = 4 KB)
    auto stage = [&](int tile, unsigned dstBase) {
        const float4* gsrc = (const float4*)(Bi + (itemBase0 + tile * 16) * 64);
        #pragma unroll
        for (int q = 0; q < 8; q++) {
            const float4* ga = gsrc + q * 32 + lane;
            const unsigned da = dstBase + (unsigned)(q * 512 + lane * 16);
            asm volatile("global_load_async_to_lds_b128 %0, %1, off"
                         :: "v"(da), "v"(ga) : "memory");
        }
    };

    stage(0, biBase0);  // prologue

    for (int j = 0; j < 16; j++) {
        if (j < 15) {
            stage(j + 1, (j & 1) ? biBase0 : biBase1);     // prefetch next tile
            asm volatile("s_wait_asynccnt 8" ::: "memory"); // tile j landed (in-order)
        } else {
            asm volatile("s_wait_asynccnt 0" ::: "memory");
        }

        const int itemBase = itemBase0 + j * 16;
        const float* ldsB = ldsBi[wave][j & 1];
        float* ldsH = ldsH2[wave];

        // ---- layer1 (elementwise) fused into A-fragment build; GEMM1 ----
        v8f acc0 = {}; v8f acc1 = {};
        #pragma unroll
        for (int c = 0; c < 2; c++) {
            v16h a;
            #pragma unroll
            for (int t = 0; t < 16; t++) {
                const int k = c * 32 + ((t >> 3) << 4) + (hi << 3) + (t & 7);
                const float h1 = av[c * 16 + t] + ldsB[ln * 64 + k];
                a[t] = (_Float16)fmaxf(h1, 0.f);
            }
            acc0 = __builtin_amdgcn_wmma_f32_16x16x32_f16(false, a, false, bf[c * 2 + 0],
                                                          (short)0, acc0, false, false);
            acc1 = __builtin_amdgcn_wmma_f32_16x16x32_f16(false, a, false, bf[c * 2 + 1],
                                                          (short)0, acc1, false, false);
        }

        // ---- bias+relu, spill H2 (C/D layout: M = r + 8*hi, N = ln) to LDS ----
        #pragma unroll
        for (int r = 0; r < 8; r++) {
            const int m = r + 8 * hi;
            ldsH[m * 32 + ln]      = fmaxf(acc0[r] + bias2a, 0.f);
            ldsH[m * 32 + 16 + ln] = fmaxf(acc1[r] + bias2b, 0.f);
        }
        // wave-private LDS: hardware DS ops are in-order per wave; fence the
        // compiler scheduler only (no cross-wave s_barrier needed).
        __builtin_amdgcn_wave_barrier();

        // ---- GEMM2: A = H2 (16x32) reloaded in A layout ----
        v16h a2;
        #pragma unroll
        for (int t = 0; t < 16; t++) {
            const int k = ((t >> 3) << 4) + (hi << 3) + (t & 7);
            a2[t] = (_Float16)ldsH[ln * 32 + k];
        }
        v8f acc3 = {};
        acc3 = __builtin_amdgcn_wmma_f32_16x16x32_f16(false, a2, false, b4f,
                                                      (short)0, acc3, false, false);

        // ---- h3 = relu(acc3 + b4); dot with Wo[128:144]; 16-lane xor reduce ----
        #pragma unroll
        for (int r = 0; r < 8; r++) {
            float p = fmaxf(acc3[r] + bias4, 0.f) * woh;
            p += __shfl_xor(p, 1);
            p += __shfl_xor(p, 2);
            p += __shfl_xor(p, 4);
            p += __shfl_xor(p, 8);
            if (ln == r) {
                const int m = r + 8 * hi;          // pair row in tile
                const int item = itemBase + m;
                out[u * 1024 + item] = fmaxf(p + su + Si[item] + bov, 0.f);
            }
        }
        __builtin_amdgcn_wave_barrier();  // order H2 reads before next-iter writes
    }
}

// ---------------------------------------------------------------------------
extern "C" void kernel_launch(void* const* d_in, const int* in_sizes, int n_in,
                              void* d_out, int out_size, void* d_ws, size_t ws_size,
                              hipStream_t stream) {
    const int*   user_id      = (const int*)  d_in[0];
    const int*   item_id      = (const int*)  d_in[1];
    const float* user_emb_mf  = (const float*)d_in[2];
    const float* item_emb_mf  = (const float*)d_in[3];
    const float* user_emb_mlp = (const float*)d_in[4];
    const float* item_emb_mlp = (const float*)d_in[5];
    const float* W1 = (const float*)d_in[6];
    const float* b1 = (const float*)d_in[7];
    const float* W2 = (const float*)d_in[8];
    const float* b2 = (const float*)d_in[9];
    const float* W4 = (const float*)d_in[10];
    const float* b4 = (const float*)d_in[11];
    const float* Wo = (const float*)d_in[12];
    const float* bo = (const float*)d_in[13];
    float* out = (float*)d_out;

    // workspace layout (~400 KB)
    float* Au = (float*)d_ws;           // 512 x 64
    float* Bi = Au + NUSER * 64;        // 1024 x 64
    float* Su = Bi + NITEM * 64;        // 512
    float* Si = Su + NUSER;             // 1024

    nmf_precompute<<<NUSER + NITEM, 64, 0, stream>>>(
        user_id, item_id, user_emb_mf, item_emb_mf, user_emb_mlp, item_emb_mlp,
        W1, Wo, Au, Bi, Su, Si);

    nmf_main<<<NUSER, 128, 0, stream>>>(
        Au, Bi, Su, Si, b1, W2, b2, W4, b4, Wo, bo, out);
}
